// MultiHeadCrossAttention_62002147885480
// MI455X (gfx1250) — matrix operand order repaired and checked by reading, not compiled, not run on hardware
//
#include <hip/hip_runtime.h>

typedef __attribute__((ext_vector_type(16))) _Float16 v16h;
typedef __attribute__((ext_vector_type(8)))  _Float16 v8h;
typedef __attribute__((ext_vector_type(4)))  _Float16 v4h;
typedef __attribute__((ext_vector_type(8)))  float    v8f;

#define D_MODEL 1024
#define SEQ     1024
#define NB      4
#define NH      16
#define HD      64
#define MD      (4096 * 1024)      // activation elements
#define WN      (1024 * 1024)      // weight elements
#define CHUNK   8                  // (b,h) pairs per score-buffer pass

// ---------------------------------------------------------------------------
// WMMA fragment loaders (16x16x32 f16, wave32) for C = A (row-major,
// K-contig) x W^T (W row-major [N,K], K-contig).
//   A frag: lane t -> row m=t&15, g=t>>4; halfs[0..7]=A[m][kb+g*8..+8),
//           halfs[8..15]=A[m][kb+16+g*8..+8)      (2 x 16B contiguous)
//   B frag: lane t -> col n=t&15; halfs[0..7]=W[n][kb+g*8..+8),
//           halfs[8..15]=W[n][kb+16+g*8..+8)      (2 x 16B contiguous)
//   C/D   : elem r -> row r+8*(t>>4), col t&15
// ---------------------------------------------------------------------------
__device__ __forceinline__ v16h load_a_frag(const _Float16* __restrict__ A,
                                            int lda, int kb, int lane) {
    const int m = lane & 15, g = lane >> 4;
    const _Float16* base = A + (size_t)m * lda + kb + g * 8;
    v8h lo = *(const v8h*)(base);
    v8h hi = *(const v8h*)(base + 16);
    v16h r;
#pragma unroll
    for (int i = 0; i < 8; ++i) { r[i] = lo[i]; r[i + 8] = hi[i]; }
    return r;
}

__device__ __forceinline__ v16h load_b_frag(const _Float16* __restrict__ Wm,
                                            int ldb, int kb, int lane) {
    const int n = lane & 15, g = lane >> 4;
    const _Float16* base = Wm + (size_t)n * ldb + kb + g * 8;
    v8h lo = *(const v8h*)(base);
    v8h hi = *(const v8h*)(base + 16);
    v16h r;
#pragma unroll
    for (int i = 0; i < 8; ++i) { r[i] = lo[i]; r[i + 8] = hi[i]; }
    return r;
}

__device__ __forceinline__ v8f wmma_f16(v16h a, v16h b, v8f c) {
    return __builtin_amdgcn_wmma_f32_16x16x32_f16(false, a, false, b,
                                                  (short)0, c, false, false);
}

// ---------------------------------------------------------------------------
// Async global->LDS copy of one k-step tile: A 64x32 halfs + B 128x32 halfs.
// 256 threads, 16B per async op, 3 ops/thread.  Tracked by ASYNCcnt.
// ---------------------------------------------------------------------------
#define A_HALFS  (64 * 32)
#define B_HALFS  (128 * 32)
#define BUF_HALFS (A_HALFS + B_HALFS)   // 6144 halfs = 12KB

__device__ __forceinline__ void async_ld_b128(_Float16* lptr, const _Float16* gptr) {
    unsigned laddr = (unsigned)(uintptr_t)lptr;     // low 32 bits = LDS offset
    unsigned long long gaddr = (unsigned long long)(uintptr_t)gptr;
    asm volatile("global_load_async_to_lds_b128 %0, %1, off"
                 :: "v"(laddr), "v"(gaddr) : "memory");
}

__device__ __forceinline__ void async_tile_copy(const _Float16* __restrict__ gA,
                                                const _Float16* __restrict__ gB,
                                                int ld, int kb,
                                                _Float16* lbuf, int tid) {
    const int row = tid >> 2;       // 0..63
    const int ch  = tid & 3;        // 4 x 8 halfs = 32 halfs per row
    async_ld_b128(lbuf + row * 32 + ch * 8,
                  gA + (size_t)row * ld + kb + ch * 8);
#pragma unroll
    for (int u = 0; u < 2; ++u) {
        const int n = row + 64 * u;
        async_ld_b128(lbuf + A_HALFS + n * 32 + ch * 8,
                      gB + (size_t)n * ld + kb + ch * 8);
    }
}

__device__ __forceinline__ void wait_async_all() {
    asm volatile("s_wait_asynccnt 0x0" ::: "memory");
}

// ---------------------------------------------------------------------------
// f32 -> f16 convert, 4 elems/thread (b128 in, b64 out)
// ---------------------------------------------------------------------------
__global__ __launch_bounds__(256)
void cvt_f16_k(const float* __restrict__ src, _Float16* __restrict__ dst, int n) {
    int i = (blockIdx.x * 256 + threadIdx.x) * 4;
    if (i < n) {
        float4 v = *(const float4*)(src + i);
        v4h o = {(_Float16)v.x, (_Float16)v.y, (_Float16)v.z, (_Float16)v.w};
        *(v4h*)(dst + i) = o;
    }
}

// ---------------------------------------------------------------------------
// Big GEMM: out = A[4096,1024] @ W[1024,1024]^T + bias.
// 8 waves (4M x 2N), block tile 64x128, async double-buffered LDS staging.
// mode 0: f16 row-major   mode 1: f16 V^T per head  mode 2: f32 row-major
// ---------------------------------------------------------------------------
__global__ __launch_bounds__(256)
void gemm_xwt_k(const _Float16* __restrict__ A, const _Float16* __restrict__ W,
                const float* __restrict__ bias, void* __restrict__ out, int mode) {
    const int K = 1024;
    __shared__ _Float16 lds[2 * BUF_HALFS];
    const int tid = threadIdx.x;
    const int lane = tid & 31, wave = tid >> 5;
    const int wm = wave & 3, wn = wave >> 2;
    const int row0 = blockIdx.x * 64;
    const int n0   = blockIdx.y * 128;
    const _Float16* gA = A + (size_t)row0 * K;
    const _Float16* gB = W + (size_t)n0 * K;

    async_tile_copy(gA, gB, K, 0, lds, tid);
    wait_async_all();
    __syncthreads();

    v8f z = {};
    v8f acc[4] = {z, z, z, z};
    for (int kb = 0; kb < K; kb += 32) {
        const int buf = (kb >> 5) & 1;
        const int kbn = (kb + 32) & (K - 1);          // wraps on last iter (harmless)
        async_tile_copy(gA, gB, K, kbn, lds + (buf ^ 1) * BUF_HALFS, tid);

        const _Float16* Abuf = lds + buf * BUF_HALFS + wm * 16 * 32;
        const _Float16* Bbuf = lds + buf * BUF_HALFS + A_HALFS + wn * 64 * 32;
        v16h a = load_a_frag(Abuf, 32, 0, lane);
#pragma unroll
        for (int t = 0; t < 4; ++t) {
            v16h b = load_b_frag(Bbuf + t * 16 * 32, 32, 0, lane);
            acc[t] = wmma_f16(a, b, acc[t]);
        }
        wait_async_all();       // own prefetch done (overlapped with WMMAs)
        __syncthreads();        // everyone done: next buf full, this buf reusable
    }

    const int g = lane >> 4, nl = lane & 15;
#pragma unroll
    for (int t = 0; t < 4; ++t) {
        const int col = n0 + (wn * 64) + 16 * t + nl;
        const float bb = bias[col];
#pragma unroll
        for (int r = 0; r < 8; ++r) {
            const int row = row0 + wm * 16 + g * 8 + r;
            const float v = acc[t][r] + bb;
            if (mode == 0) {
                ((_Float16*)out)[(size_t)row * 1024 + col] = (_Float16)v;
            } else if (mode == 1) {   // V^T: vt[(b*16+h)*64 + d][s]
                const int b_ = row >> 10, s_ = row & 1023;
                const int h_ = col >> 6,  d_ = col & 63;
                ((_Float16*)out)[(((size_t)(b_ * 16 + h_) * 64 + d_) << 10) + s_] =
                    (_Float16)v;
            } else {
                ((float*)out)[(size_t)row * 1024 + col] = v;
            }
        }
    }
}

// ---------------------------------------------------------------------------
// Scores: S[i][j] = (1/8) * sum_d q[i][d]*k[j][d] per (b,h), K=64.
// All 10 fragments loaded up front, then 8 WMMAs.  grid (16, 8, CHUNK)
// ---------------------------------------------------------------------------
__global__ __launch_bounds__(256)
void scores_k(const _Float16* __restrict__ q16, const _Float16* __restrict__ k16,
              float* __restrict__ scores, int bh0) {
    const int bh = bh0 + blockIdx.z;
    const int b = bh >> 4, h = bh & 15;
    const _Float16* Aq = q16 + (size_t)b * SEQ * D_MODEL + h * 64;
    const _Float16* Bk = k16 + (size_t)b * SEQ * D_MODEL + h * 64;
    float* Sc = scores + (size_t)blockIdx.z * SEQ * SEQ;

    const int lane = threadIdx.x & 31, wave = threadIdx.x >> 5;
    const int row0 = blockIdx.x * 64 + (wave & 3) * 16;
    const int n0   = blockIdx.y * 128 + (wave >> 2) * 64;

    const _Float16* Arow = Aq + (size_t)row0 * D_MODEL;
    v16h a0 = load_a_frag(Arow, D_MODEL, 0,  lane);
    v16h a1 = load_a_frag(Arow, D_MODEL, 32, lane);
    v16h bf[8];
#pragma unroll
    for (int t = 0; t < 4; ++t) {
        const _Float16* Brow = Bk + (size_t)(n0 + 16 * t) * D_MODEL;
        bf[t]     = load_b_frag(Brow, D_MODEL, 0,  lane);
        bf[t + 4] = load_b_frag(Brow, D_MODEL, 32, lane);
    }
    v8f z = {};
    v8f acc[4] = {z, z, z, z};
#pragma unroll
    for (int t = 0; t < 4; ++t) acc[t] = wmma_f16(a0, bf[t], acc[t]);
#pragma unroll
    for (int t = 0; t < 4; ++t) acc[t] = wmma_f16(a1, bf[t + 4], acc[t]);

    const int g = lane >> 4, nl = lane & 15;
#pragma unroll
    for (int t = 0; t < 4; ++t) {
        const int col = n0 + 16 * t + nl;
#pragma unroll
        for (int r = 0; r < 8; ++r)
            Sc[(size_t)(row0 + g * 8 + r) * SEQ + col] = acc[t][r] * 0.125f;
    }
}

// ---------------------------------------------------------------------------
// Row softmax: P_q[i][:] = softmax_j(S[i][:]).  grid (1024, CHUNK)
// ---------------------------------------------------------------------------
__global__ __launch_bounds__(256)
void softmax_row_k(const float* __restrict__ scores, _Float16* __restrict__ P) {
    const int tid = threadIdx.x;
    const float* Sr = scores + (size_t)blockIdx.y * SEQ * SEQ + (size_t)blockIdx.x * SEQ;
    _Float16* Pr = P + (size_t)blockIdx.y * SEQ * SEQ + (size_t)blockIdx.x * SEQ;
    __shared__ float red[256];

    float v[4];
#pragma unroll
    for (int u = 0; u < 4; ++u) v[u] = Sr[tid + 256 * u];
    float m = fmaxf(fmaxf(v[0], v[1]), fmaxf(v[2], v[3]));
    red[tid] = m; __syncthreads();
    for (int s = 128; s > 0; s >>= 1) {
        if (tid < s) red[tid] = fmaxf(red[tid], red[tid + s]);
        __syncthreads();
    }
    m = red[0]; __syncthreads();

    float e[4], sum = 0.f;
#pragma unroll
    for (int u = 0; u < 4; ++u) { e[u] = __expf(v[u] - m); sum += e[u]; }
    red[tid] = sum; __syncthreads();
    for (int s = 128; s > 0; s >>= 1) {
        if (tid < s) red[tid] += red[tid + s];
        __syncthreads();
    }
    const float r = 1.0f / red[0];
#pragma unroll
    for (int u = 0; u < 4; ++u) Pr[tid + 256 * u] = (_Float16)(e[u] * r);
}

// ---------------------------------------------------------------------------
// Column softmax, transposed store: P_k[j][i] = exp(S[i][j]-max_i)/sum_i.
// grid (32, CHUNK), block (32,8), LDS 32x32 transpose for coalesced writes.
// ---------------------------------------------------------------------------
__global__ void softmax_colT_k(const float* __restrict__ scores,
                               _Float16* __restrict__ P) {
    const int tx = threadIdx.x, ty = threadIdx.y;
    const int j0 = blockIdx.x * 32;
    const int j  = j0 + tx;
    const float* Sc = scores + (size_t)blockIdx.y * SEQ * SEQ;
    _Float16* Pc = P + (size_t)blockIdx.y * SEQ * SEQ;

    __shared__ float red[8][32];
    __shared__ float cstat[32];

    float m = -3.4e38f;
    for (int i = ty; i < SEQ; i += 8) m = fmaxf(m, Sc[(size_t)i * SEQ + j]);
    red[ty][tx] = m; __syncthreads();
    if (ty == 0) {
        float mm = red[0][tx];
#pragma unroll
        for (int r = 1; r < 8; ++r) mm = fmaxf(mm, red[r][tx]);
        cstat[tx] = mm;
    }
    __syncthreads();
    const float cm = cstat[tx];
    __syncthreads();

    float s = 0.f;
    for (int i = ty; i < SEQ; i += 8) s += __expf(Sc[(size_t)i * SEQ + j] - cm);
    red[ty][tx] = s; __syncthreads();
    if (ty == 0) {
        float ss = red[0][tx];
#pragma unroll
        for (int r = 1; r < 8; ++r) ss += red[r][tx];
        cstat[tx] = 1.0f / ss;
    }
    __syncthreads();
    const float rs = cstat[tx];

    __shared__ float tile[32][33];
    for (int i0 = 0; i0 < SEQ; i0 += 32) {
        for (int ii = ty; ii < 32; ii += 8)
            tile[ii][tx] = __expf(Sc[(size_t)(i0 + ii) * SEQ + j] - cm) * rs;
        __syncthreads();
        for (int jj = ty; jj < 32; jj += 8)
            Pc[(size_t)(j0 + jj) * SEQ + (i0 + tx)] = (_Float16)tile[tx][jj];
        __syncthreads();
    }
}

// ---------------------------------------------------------------------------
// X = P[1024,1024] @ V (via V^T [64,1024]).  Register double-buffered K loop.
// grid (8, CHUNK, 2): z=0 -> P_q -> xq, z=1 -> P_k -> xk.
// ---------------------------------------------------------------------------
__global__ __launch_bounds__(256)
void pv_k(const _Float16* __restrict__ Pq, const _Float16* __restrict__ Pk,
          const _Float16* __restrict__ vt, _Float16* __restrict__ xq,
          _Float16* __restrict__ xk, int bh0) {
    const int c = blockIdx.y;
    const int bh = bh0 + c;
    const int b = bh >> 4, h = bh & 15;
    const _Float16* Ap = (blockIdx.z ? Pk : Pq) + (size_t)c * SEQ * SEQ;
    _Float16* out16 = blockIdx.z ? xk : xq;
    const _Float16* Bv = vt + (size_t)bh * 64 * SEQ;

    const int lane = threadIdx.x & 31, wave = threadIdx.x >> 5;
    const int row0 = blockIdx.x * 128 + wave * 16;
    const _Float16* Arow = Ap + (size_t)row0 * SEQ;

    v8f z = {};
    v8f acc[4] = {z, z, z, z};
    v16h a_cur = load_a_frag(Arow, SEQ, 0, lane);
    v16h b_cur[4];
#pragma unroll
    for (int t = 0; t < 4; ++t)
        b_cur[t] = load_b_frag(Bv + (size_t)(16 * t) * SEQ, SEQ, 0, lane);

    for (int kb = 0; kb < SEQ; kb += 32) {
        const int kbn = (kb + 32) & (SEQ - 1);        // wraps on last iter
        v16h a_nxt = load_a_frag(Arow, SEQ, kbn, lane);
        v16h b_nxt[4];
#pragma unroll
        for (int t = 0; t < 4; ++t)
            b_nxt[t] = load_b_frag(Bv + (size_t)(16 * t) * SEQ, SEQ, kbn, lane);
#pragma unroll
        for (int t = 0; t < 4; ++t) acc[t] = wmma_f16(a_cur, b_cur[t], acc[t]);
        a_cur = a_nxt;
#pragma unroll
        for (int t = 0; t < 4; ++t) b_cur[t] = b_nxt[t];
    }

    const int g = lane >> 4, nl = lane & 15;
#pragma unroll
    for (int t = 0; t < 4; ++t) {
        const int d = 16 * t + nl;
#pragma unroll
        for (int r = 0; r < 8; ++r) {
            const int row = row0 + g * 8 + r;
            out16[((size_t)b * SEQ + row) * D_MODEL + h * 64 + d] = (_Float16)acc[t][r];
        }
    }
}

// ---------------------------------------------------------------------------
// LayerNorm(residual + pre) * gamma + beta.  grid (4096)
// ---------------------------------------------------------------------------
__global__ __launch_bounds__(256)
void ln_k(const float* __restrict__ pre, const float* __restrict__ resid,
          const float* __restrict__ gamma, const float* __restrict__ beta,
          float* __restrict__ out) {
    const int tid = threadIdx.x;
    const size_t base = (size_t)blockIdx.x * 1024;
    __shared__ float red[256];

    float x[4];
#pragma unroll
    for (int u = 0; u < 4; ++u)
        x[u] = pre[base + tid + 256 * u] + resid[base + tid + 256 * u];

    float s = x[0] + x[1] + x[2] + x[3];
    red[tid] = s; __syncthreads();
    for (int st = 128; st > 0; st >>= 1) {
        if (tid < st) red[tid] += red[tid + st];
        __syncthreads();
    }
    const float mu = red[0] * (1.0f / 1024.0f);
    __syncthreads();

    float v = 0.f;
#pragma unroll
    for (int u = 0; u < 4; ++u) { float d = x[u] - mu; v += d * d; }
    red[tid] = v; __syncthreads();
    for (int st = 128; st > 0; st >>= 1) {
        if (tid < st) red[tid] += red[tid + st];
        __syncthreads();
    }
    const float inv = rsqrtf(red[0] * (1.0f / 1024.0f) + 1e-5f);
    __syncthreads();
#pragma unroll
    for (int u = 0; u < 4; ++u) {
        const int col = tid + 256 * u;
        out[base + col] = (x[u] - mu) * inv * gamma[col] + beta[col];
    }
}

// ---------------------------------------------------------------------------
extern "C" void kernel_launch(void* const* d_in, const int* in_sizes, int n_in,
                              void* d_out, int out_size, void* d_ws, size_t ws_size,
                              hipStream_t stream) {
    const float* query = (const float*)d_in[0];
    const float* key   = (const float*)d_in[1];
    const float* value = (const float*)d_in[2];
    const float* Wq  = (const float*)d_in[3];  const float* bq  = (const float*)d_in[4];
    const float* Wk  = (const float*)d_in[5];  const float* bk  = (const float*)d_in[6];
    const float* Wv  = (const float*)d_in[7];  const float* bv  = (const float*)d_in[8];
    const float* Wfq = (const float*)d_in[9];  const float* bfq = (const float*)d_in[10];
    const float* Wfk = (const float*)d_in[11]; const float* bfk = (const float*)d_in[12];
    const float* gq  = (const float*)d_in[13]; const float* btq = (const float*)d_in[14];
    const float* gk  = (const float*)d_in[15]; const float* btk = (const float*)d_in[16];
    float* out = (float*)d_out;

    // ---- workspace carve-up ----
    char* w = (char*)d_ws;
    size_t off = 0;
    auto takeH = [&](size_t elems) { _Float16* p = (_Float16*)(w + off); off += elems * 2; return p; };
    auto takeF = [&](size_t elems) { float* p = (float*)(w + off); off += elems * 4; return p; };

    _Float16* qry16 = takeH(MD);
    _Float16* key16 = takeH(MD);
    _Float16* val16 = takeH(MD);
    _Float16* Wq16  = takeH(WN);
    _Float16* Wk16  = takeH(WN);
    _Float16* Wv16  = takeH(WN);
    _Float16* Wfq16 = takeH(WN);
    _Float16* Wfk16 = takeH(WN);
    _Float16* q16   = takeH(MD);
    _Float16* k16   = takeH(MD);
    _Float16* vt16  = takeH(MD);
    _Float16* xq16  = takeH(MD);
    _Float16* xk16  = takeH(MD);
    float*    preq  = takeF(MD);
    float*    prek  = takeF(MD);
    float*    scoreB = takeF((size_t)CHUNK * SEQ * SEQ);
    _Float16* Pq16  = takeH((size_t)CHUNK * SEQ * SEQ);
    _Float16* Pk16  = takeH((size_t)CHUNK * SEQ * SEQ);
    if (off > ws_size) return;

    // ---- 1) f32 -> f16 conversions ----
    cvt_f16_k<<<MD / 1024, 256, 0, stream>>>(query, qry16, MD);
    cvt_f16_k<<<MD / 1024, 256, 0, stream>>>(key,   key16, MD);
    cvt_f16_k<<<MD / 1024, 256, 0, stream>>>(value, val16, MD);
    cvt_f16_k<<<WN / 1024, 256, 0, stream>>>(Wq,  Wq16,  WN);
    cvt_f16_k<<<WN / 1024, 256, 0, stream>>>(Wk,  Wk16,  WN);
    cvt_f16_k<<<WN / 1024, 256, 0, stream>>>(Wv,  Wv16,  WN);
    cvt_f16_k<<<WN / 1024, 256, 0, stream>>>(Wfq, Wfq16, WN);
    cvt_f16_k<<<WN / 1024, 256, 0, stream>>>(Wfk, Wfk16, WN);

    // ---- 2) Q/K/V projections (WMMA + async-LDS pipeline) ----
    dim3 gemmGrid(4096 / 64, 1024 / 128);
    gemm_xwt_k<<<gemmGrid, 256, 0, stream>>>(qry16, Wq16, bq, q16,  0);
    gemm_xwt_k<<<gemmGrid, 256, 0, stream>>>(key16, Wk16, bk, k16,  0);
    gemm_xwt_k<<<gemmGrid, 256, 0, stream>>>(val16, Wv16, bv, vt16, 1);

    // ---- 3) attention, CHUNK (b,h) pairs per pass ----
    for (int bh0 = 0; bh0 < NB * NH; bh0 += CHUNK) {
        scores_k<<<dim3(16, 8, CHUNK), 256, 0, stream>>>(q16, k16, scoreB, bh0);
        softmax_row_k<<<dim3(SEQ, CHUNK), 256, 0, stream>>>(scoreB, Pq16);
        softmax_colT_k<<<dim3(32, CHUNK), dim3(32, 8), 0, stream>>>(scoreB, Pk16);
        pv_k<<<dim3(8, CHUNK, 2), 256, 0, stream>>>(Pq16, Pk16, vt16, xq16, xk16, bh0);
    }

    // ---- 4) output projections (WMMA, f32 out) ----
    gemm_xwt_k<<<gemmGrid, 256, 0, stream>>>(xq16, Wfq16, bfq, preq, 2);
    gemm_xwt_k<<<gemmGrid, 256, 0, stream>>>(xk16, Wfk16, bfk, prek, 2);

    // ---- 5) residual + LayerNorm -> d_out ----
    ln_k<<<4096, 256, 0, stream>>>(preq, query, gq, btq, out);
    ln_k<<<4096, 256, 0, stream>>>(prek, key,   gk, btk, out + MD);
}
